// GLADEncoder_global_local_no_rnn_v2_65395172049608
// MI455X (gfx1250) — matrix operand order repaired and checked by reading, not compiled, not run on hardware
//
#include <hip/hip_runtime.h>
#include <hip/hip_bf16.h>
#include <math.h>

typedef __attribute__((ext_vector_type(16))) _Float16 v16h;
typedef __attribute__((ext_vector_type(8)))  float    v8f;
typedef __attribute__((ext_vector_type(2)))  float    v2f;

#if defined(__has_builtin)
#if __has_builtin(__builtin_amdgcn_wmma_f32_16x16x4_f32)
#define HAVE_WMMA_F32X4 1
#endif
#endif

constexpr int BB = 32;    // batch
constexpr int SS = 1024;  // seq len
constexpr int DD = 512;   // feature dim

// ---------------------------------------------------------------- utilities
__global__ void k_zero(float* __restrict__ p, int n) {
  int i = blockIdx.x * blockDim.x + threadIdx.x;
  if (i < n) p[i] = 0.f;
}

// -------------------------------------------- pass A: copy x->out + colsum
// grid: BB * (SS/128), block: 128 threads (4 waves); thread t owns cols 4t..4t+3
__global__ void k_copy_colsum(const float* __restrict__ x, float* __restrict__ out,
                              float* __restrict__ xsum) {
  constexpr int ROWS = 128;
  const int chunks = SS / ROWS;
  const int b  = blockIdx.x / chunks;
  const int r0 = (blockIdx.x % chunks) * ROWS;
  const int c4 = threadIdx.x * 4;
  const float4* xr = (const float4*)(x   + ((size_t)b * SS + r0) * DD + c4);
  float4*       orw = (float4*)(out + ((size_t)b * SS + r0) * DD + c4);
  float4 acc = make_float4(0.f, 0.f, 0.f, 0.f);
  for (int r = 0; r < ROWS; ++r) {
    if (r + 8 < ROWS) __builtin_prefetch(&xr[(size_t)(r + 8) * (DD / 4)], 0, 0);
    float4 v = xr[(size_t)r * (DD / 4)];
    orw[(size_t)r * (DD / 4)] = v;
    acc.x += v.x; acc.y += v.y; acc.z += v.z; acc.w += v.w;
  }
  float* xs = xsum + b * DD + c4;
  atomicAdd(xs + 0, acc.x);
  atomicAdd(xs + 1, acc.y);
  atomicAdd(xs + 2, acc.z);
  atomicAdd(xs + 3, acc.w);
}

// ------------------------------------- wave-per-row dot: scores[b,s] = x.vec + bias
// vecStride: 0 (shared w_score) or DD (per-batch v). biasStride: 0 or 1.
__global__ void k_row_dot(const float* __restrict__ x,
                          const float* __restrict__ vec, int vecStride,
                          const float* __restrict__ bias, int biasStride,
                          float* __restrict__ outS) {
  const int wavesPerBlock = blockDim.x >> 5;
  const int row  = blockIdx.x * wavesPerBlock + (threadIdx.x >> 5);
  const int lane = threadIdx.x & 31;
  const int b = row / SS;
  const int s = row % SS;
  const float4* xr = (const float4*)(x + ((size_t)b * SS + s) * DD);
  const float4* vr = (const float4*)(vec + (size_t)b * vecStride);
  float p = 0.f;
  #pragma unroll
  for (int k = 0; k < DD / 128; ++k) {        // 4 iters, 512B coalesced per wave
    float4 xv = xr[lane + k * 32];
    float4 wv = vr[lane + k * 32];
    p += xv.x * wv.x + xv.y * wv.y + xv.z * wv.z + xv.w * wv.w;
  }
  #pragma unroll
  for (int off = 16; off; off >>= 1) p += __shfl_xor(p, off, 32);
  if (lane == 0) outS[row] = p + bias[b * biasStride];
}

// -------------------------------------------------- WMMA GEMM (M=32,K=512,N=512)
// C[m,n] = outScale * sum_k A[m,k]*B(k,n) + biasScale * biasN[n]
// B(k,n) = bTrans ? Bm[n*DD+k] : Bm[k*DD+n]
// One wave per 16x16 tile: 2 M-tiles x 32 N-tiles = 64 waves.
// Preferred: native fp32 V_WMMA_F32_16X16X4_F32 (exact); fallback: f16 16x16x32.
__global__ void k_gemm_wmma(const float* __restrict__ A, const float* __restrict__ Bm,
                            const float* __restrict__ biasN, float biasScale,
                            float outScale, int bTrans, float* __restrict__ Out) {
  const int wave = (blockIdx.x * blockDim.x + threadIdx.x) >> 5;
  const int lane = threadIdx.x & 31;
  const int tm = wave >> 5;     // 0..1
  const int tn = wave & 31;     // 0..31
  const int ln = lane & 15;
  const int hi = lane >> 4;     // K-subblock select per ISA layouts
  v8f acc = {0.f, 0.f, 0.f, 0.f, 0.f, 0.f, 0.f, 0.f};
  const int arow = tm * 16 + ln;            // A row (M)
  const int bcol = tn * 16 + ln;            // B col (N)

#if HAVE_WMMA_F32X4
  // f32 16x16x4: A 16x4 layout: lanes0-15 hold K=0,1 (regs 0,1); lanes16-31 K=2,3.
  // B 4x16 mirrored: lane = N, regs hold K=hi*2, hi*2+1.
  #pragma unroll 4
  for (int kb = 0; kb < DD / 4; ++kb) {
    const int ka = kb * 4 + hi * 2;
    v2f a, bv;
    a.x = A[(size_t)arow * DD + ka];
    a.y = A[(size_t)arow * DD + ka + 1];
    bv.x = bTrans ? Bm[(size_t)bcol * DD + ka]     : Bm[(size_t)ka * DD + bcol];
    bv.y = bTrans ? Bm[(size_t)bcol * DD + ka + 1] : Bm[(size_t)(ka + 1) * DD + bcol];
    acc = __builtin_amdgcn_wmma_f32_16x16x4_f32(
        /*neg_a=*/false, a, /*neg_b=*/false, bv,
        /*c_mod=*/(short)0, acc, /*reuse_a=*/false, /*reuse_b=*/false);
  }
#else
  for (int kb = 0; kb < DD / 32; ++kb) {      // 16 K-steps of 32
    v16h a, bv;
    #pragma unroll
    for (int r = 0; r < 8; ++r) {
      // A 16x32 f16 layout: VGPR r<4 -> K=2r(+8 if hi); r>=4 -> K=16+2(r-4)(+8 if hi)
      int ka = ((r & 4) ? 16 : 0) + 2 * (r & 3) + hi * 8 + kb * 32;
      a[2 * r]     = (_Float16)A[(size_t)arow * DD + ka];
      a[2 * r + 1] = (_Float16)A[(size_t)arow * DD + ka + 1];
      // B 32x16 f16 layout: same K order as A (r<4 -> K=2r(+8 if hi); r>=4 -> K=16+2(r-4)(+8 if hi))
      int kbk = ((r & 4) ? 16 : 0) + 2 * (r & 3) + hi * 8 + kb * 32;
      float b0 = bTrans ? Bm[(size_t)bcol * DD + kbk]     : Bm[(size_t)kbk * DD + bcol];
      float b1 = bTrans ? Bm[(size_t)bcol * DD + kbk + 1] : Bm[(size_t)(kbk + 1) * DD + bcol];
      bv[2 * r]     = (_Float16)b0;
      bv[2 * r + 1] = (_Float16)b1;
    }
    acc = __builtin_amdgcn_wmma_f32_16x16x32_f16(
        /*neg_a=*/false, a, /*neg_b=*/false, bv,
        /*c_mod=*/(short)0, acc, /*reuse_a=*/false, /*reuse_b=*/false);
  }
#endif

  // C/D 16x16 f32 layout: VGPR r -> M = r + 8*hi, N = ln
  #pragma unroll
  for (int r = 0; r < 8; ++r) {
    int m = tm * 16 + r + hi * 8;
    int n = tn * 16 + ln;
    Out[(size_t)m * DD + n] = outScale * acc[r] + biasScale * biasN[n];
  }
}

// ------------------------------------------------------- c0[b] = bq.ksum[b]*scale
__global__ void k_c0(const float* __restrict__ ksum, const float* __restrict__ bq,
                     float scale, float* __restrict__ c0) {
  const int wave = threadIdx.x >> 5;  // == b, block of 1024 = 32 waves
  const int lane = threadIdx.x & 31;
  float p = 0.f;
  for (int e = lane; e < DD; e += 32) p += bq[e] * ksum[wave * DD + e];
  #pragma unroll
  for (int off = 16; off; off >>= 1) p += __shfl_xor(p, off, 32);
  if (lane == 0) c0[wave] = p * scale;
}

// ---------------- masked double softmax + beta combine -> one weight vector w[b,s]
__global__ void k_softmax_combine(const float* __restrict__ ls, const float* __restrict__ gs,
                                  const int* __restrict__ x_len, const int* __restrict__ slot_idx,
                                  const float* __restrict__ beta_raw, float* __restrict__ w) {
  __shared__ float red[256];
  const int b = blockIdx.x, t = threadIdx.x;
  const int len = x_len[b];                       // 1..SS guaranteed
  const float beta = 1.f / (1.f + expf(-beta_raw[slot_idx[0]]));
  const float* lrow = ls + (size_t)b * SS;
  const float* grow = gs + (size_t)b * SS;

  float lm = -INFINITY, gm = -INFINITY;
  for (int s = t; s < len; s += 256) { lm = fmaxf(lm, lrow[s]); gm = fmaxf(gm, grow[s]); }
  red[t] = lm; __syncthreads();
  for (int o = 128; o; o >>= 1) { if (t < o) red[t] = fmaxf(red[t], red[t + o]); __syncthreads(); }
  lm = red[0]; __syncthreads();
  red[t] = gm; __syncthreads();
  for (int o = 128; o; o >>= 1) { if (t < o) red[t] = fmaxf(red[t], red[t + o]); __syncthreads(); }
  gm = red[0]; __syncthreads();

  float lsum = 0.f, gsum = 0.f;
  for (int s = t; s < len; s += 256) { lsum += expf(lrow[s] - lm); gsum += expf(grow[s] - gm); }
  red[t] = lsum; __syncthreads();
  for (int o = 128; o; o >>= 1) { if (t < o) red[t] += red[t + o]; __syncthreads(); }
  lsum = red[0]; __syncthreads();
  red[t] = gsum; __syncthreads();
  for (int o = 128; o; o >>= 1) { if (t < o) red[t] += red[t + o]; __syncthreads(); }
  gsum = red[0]; __syncthreads();

  const float li = beta / lsum, gi = (1.f - beta) / gsum;
  for (int s = t; s < SS; s += 256) {
    float wv = 0.f;
    if (s < len) wv = li * expf(lrow[s] - lm) + gi * expf(grow[s] - gm);
    w[(size_t)b * SS + s] = wv;
  }
}

// ------------------------------------------ c[b,d] = sum_s w[b,s] * x[b,s,d]
__global__ void k_weighted_sum(const float* __restrict__ x, const float* __restrict__ w,
                               float* __restrict__ outc) {
  const int b = blockIdx.x;
  const int c4 = threadIdx.x * 4;   // block 128 -> cols 0..511
  const float4* xr = (const float4*)(x + (size_t)b * SS * DD + c4);
  const float*  wr = w + (size_t)b * SS;
  float4 acc = make_float4(0.f, 0.f, 0.f, 0.f);
  for (int s = 0; s < SS; ++s) {
    if (s + 8 < SS) __builtin_prefetch(&xr[(size_t)(s + 8) * (DD / 4)], 0, 0);
    float wv = wr[s];
    float4 xv = xr[(size_t)s * (DD / 4)];
    acc.x += wv * xv.x; acc.y += wv * xv.y; acc.z += wv * xv.z; acc.w += wv * xv.w;
  }
  float4* o = (float4*)(outc + (size_t)b * DD + c4);
  *o = acc;
}

// ---------------------------------------------------------------------------
extern "C" void kernel_launch(void* const* d_in, const int* in_sizes, int n_in,
                              void* d_out, int out_size, void* d_ws, size_t ws_size,
                              hipStream_t stream) {
  const float* x        = (const float*)d_in[0];
  const int*   x_len    = (const int*)  d_in[1];
  const int*   slot_idx = (const int*)  d_in[2];
  const float* w_score  = (const float*)d_in[3];
  const float* b_score  = (const float*)d_in[4];
  const float* Wq       = (const float*)d_in[5];
  const float* bq       = (const float*)d_in[6];
  const float* Wk       = (const float*)d_in[7];
  const float* bk       = (const float*)d_in[8];
  const float* beta_raw = (const float*)d_in[9];

  float* out   = (float*)d_out;                        // x passthrough [B,S,D]
  float* out_c = out + (size_t)BB * SS * DD;           // c [B,D]

  float* wsf    = (float*)d_ws;
  float* ws_xsum = wsf;                                // [B,D]   16384
  float* ws_ls   = ws_xsum + BB * DD;                  // [B,S]   32768
  float* ws_gs   = ws_ls   + BB * SS;                  // [B,S]   32768
  float* ws_ksum = ws_gs   + BB * SS;                  // [B,D]   16384
  float* ws_v    = ws_ksum + BB * DD;                  // [B,D]   16384
  float* ws_c0   = ws_v    + BB * DD;                  // [B]     32
  float* ws_w    = ws_c0   + BB;                       // [B,S]   32768

  const float inv_sqrt_d = rsqrtf((float)DD);

  // 0) zero xsum accumulator
  k_zero<<<(BB * DD + 255) / 256, 256, 0, stream>>>(ws_xsum, BB * DD);

  // 1) copy x -> out, accumulate per-batch column sums
  k_copy_colsum<<<BB * (SS / 128), 128, 0, stream>>>(x, out, ws_xsum);

  // 2) local scores: ls[b,s] = x[b,s].w_score + b_score
  k_row_dot<<<(BB * SS) / 8, 256, 0, stream>>>(x, w_score, 0, b_score, 0, ws_ls);

  // 3) ksum = xsum @ Wk + S*bk                     (WMMA)
  k_gemm_wmma<<<8, 256, 0, stream>>>(ws_xsum, Wk, bk, (float)SS, 1.f, 0, ws_ksum);

  // 4) v = (ksum @ Wq^T) / sqrt(D)                 (WMMA)
  k_gemm_wmma<<<8, 256, 0, stream>>>(ws_ksum, Wq, bq, 0.f, inv_sqrt_d, 1, ws_v);

  // 4b) c0[b] = bq . ksum[b] / sqrt(D)
  k_c0<<<1, 1024, 0, stream>>>(ws_ksum, bq, inv_sqrt_d, ws_c0);

  // 5) global scores: gs[b,s] = x[b,s].v[b] + c0[b]
  k_row_dot<<<(BB * SS) / 8, 256, 0, stream>>>(x, ws_v, DD, ws_c0, 1, ws_gs);

  // 6) masked softmaxes + beta combine -> single weight vector
  k_softmax_combine<<<BB, 256, 0, stream>>>(ws_ls, ws_gs, x_len, slot_idx, beta_raw, ws_w);

  // 7) c[b,:] = sum_s w[b,s] * x[b,s,:]
  k_weighted_sum<<<BB, 128, 0, stream>>>(x, ws_w, out_c);

  (void)in_sizes; (void)n_in; (void)out_size; (void)ws_size;
}
